// HanoiMamba_82832739270879
// MI455X (gfx1250) — matrix operand order repaired and checked by reading, not compiled, not run on hardware
//
#include <hip/hip_runtime.h>
#include <hip/hip_bf16.h>

typedef _Float16 h16v __attribute__((ext_vector_type(16)));
typedef _Float16 h8v  __attribute__((ext_vector_type(8)));
typedef float    f8v  __attribute__((ext_vector_type(8)));

#define BB    16
#define LL    2048
#define NTOK  (BB*LL)      // 32768
#define DM    64
#define DSTT  16
#define HD    16
#define DIN   128
#define NH    8
#define XBCW  160
#define INW   296          // 2*DIN + 2*DSTT + NH

// ---------------------------------------------------------------------------
// Pack an f32 row-major (K x N) weight matrix into the WMMA-B f16 layout:
// Bp[((kc*NT+nt)*32+lane)*16+j] = W[kc*32 + (lane<16?0:8) + (j&7) + (j&8?16:0)][nt*16 + (lane&15)]
// (16-bit B operand: lanes 0-15 hold K=0..7,16..23, lanes 16-31 hold K=8..15,24..31)
// ---------------------------------------------------------------------------
__global__ void pack_w_f16(const float* __restrict__ W, _Float16* __restrict__ out,
                           int K, int N, int KC, int NT) {
    int t = blockIdx.x * blockDim.x + threadIdx.x;
    int total = KC * NT * 32 * 16;
    if (t >= total) return;
    int j    = t & 15;
    int lane = (t >> 4) & 31;
    int tile = t >> 9;
    int nt = tile % NT;
    int kc = tile / NT;
    int n = nt * 16 + (lane & 15);
    int k = kc * 32 + ((lane < 16) ? 0 : 8) + (j & 7) + ((j & 8) ? 16 : 0);
    float v = (k < K && n < N) ? W[k * N + n] : 0.0f;
    out[t] = (_Float16)v;
}

// ---------------------------------------------------------------------------
// Embedding gather + proj_in: one block (64 threads) per token.
// ---------------------------------------------------------------------------
__global__ void embed_proj(const int* __restrict__ aggs,
                           const float* __restrict__ peg, const float* __restrict__ cnt,
                           const float* __restrict__ top,
                           const float* __restrict__ W, const float* __restrict__ bias,
                           float* __restrict__ x) {
    __shared__ float e[80];
    int tok = blockIdx.x;
    int tid = threadIdx.x;
    if (tid < 10) {
        const float* tbl;
        switch (tid) {
            case 0: case 1: case 9: tbl = peg; break;
            case 2: case 3: case 4: case 8: tbl = cnt; break;
            default: tbl = top; break;
        }
        int idx = aggs[tok * 10 + tid];
        #pragma unroll
        for (int j = 0; j < 8; ++j) e[tid * 8 + j] = tbl[idx * 8 + j];
    }
    __syncthreads();
    float acc = bias[tid];
    #pragma unroll 8
    for (int k = 0; k < 80; ++k) acc += e[k] * W[k * 64 + tid];
    x[tok * 64 + tid] = acc;
}

// ---------------------------------------------------------------------------
// RMSNorm over D=64, write f16 A-matrix. One wave per token (8 tokens/block).
// ---------------------------------------------------------------------------
__global__ void rmsnorm_f16(const float* __restrict__ x, const float* __restrict__ rw,
                            _Float16* __restrict__ xn) {
    int lane = threadIdx.x & 31;
    int tok  = blockIdx.x * (blockDim.x >> 5) + (threadIdx.x >> 5);
    const float* row = x + (size_t)tok * 64;
    float v0 = row[lane];
    float v1 = row[lane + 32];
    float ss = v0 * v0 + v1 * v1;
    #pragma unroll
    for (int o = 16; o > 0; o >>= 1) ss += __shfl_xor(ss, o, 32);
    float sc = rsqrtf(ss * (1.0f / 64.0f) + 1e-6f);
    xn[(size_t)tok * 64 + lane]      = (_Float16)(v0 * sc * rw[lane]);
    xn[(size_t)tok * 64 + lane + 32] = (_Float16)(v1 * sc * rw[lane + 32]);
}

// ---------------------------------------------------------------------------
// WMMA GEMM: C[M x Nout] = A[M x K](f16) * Bpacked(f16) (+ Res), f32 accum.
// One wave per 16-row M-tile; A fragments held in VGPRs and reused across
// all N tiles. KC (K/32) is compile-time; RES selects fused residual add.
// ---------------------------------------------------------------------------
template <int KC, bool RES>
__global__ void gemm_wmma(const _Float16* __restrict__ A, const _Float16* __restrict__ Bp,
                          float* __restrict__ C, const float* __restrict__ Res,
                          int K, int Nout, int NT) {
    int lane = threadIdx.x;
    int mt = blockIdx.x;

    const _Float16* arow = A + (size_t)(mt * 16 + (lane & 15)) * K;
    int ko = (lane < 16) ? 0 : 8;
    h16v av[KC];
    #pragma unroll
    for (int kc = 0; kc < KC; ++kc) {
        h8v a0 = *(const h8v*)(arow + kc * 32 + ko);
        h8v a1 = *(const h8v*)(arow + kc * 32 + ko + 16);
        #pragma unroll
        for (int j = 0; j < 8; ++j) { av[kc][j] = a0[j]; av[kc][j + 8] = a1[j]; }
    }

    int rbase = mt * 16 + ((lane < 16) ? 0 : 8);
    int colo  = lane & 15;
    for (int nt = 0; nt < NT; ++nt) {
        f8v acc = {};
        #pragma unroll
        for (int kc = 0; kc < KC; ++kc) {
            h16v bv = *(const h16v*)(Bp + ((size_t)(kc * NT + nt) * 32 + lane) * 16);
            acc = __builtin_amdgcn_wmma_f32_16x16x32_f16(false, av[kc], false, bv,
                                                         (short)0, acc, false, false);
        }
        int col = nt * 16 + colo;
        if (col < Nout) {
            #pragma unroll
            for (int r = 0; r < 8; ++r) {
                size_t idx = (size_t)(rbase + r) * Nout + col;
                float v = acc[r];
                if (RES) v += Res[idx];
                C[idx] = v;
            }
        }
    }
}

// ---------------------------------------------------------------------------
// Depthwise causal conv (K=4) + bias + silu on the xbc slice of zxbcdt.
// ---------------------------------------------------------------------------
__global__ void conv_silu(const float* __restrict__ zx, const float* __restrict__ cw,
                          const float* __restrict__ cb, float* __restrict__ xo) {
    int t = blockIdx.x * blockDim.x + threadIdx.x;
    if (t >= NTOK * XBCW) return;
    int c = t % XBCW;
    int tok = t / XBCW;
    int b = tok / LL, l = tok % LL;
    float acc = cb[c];
    #pragma unroll
    for (int k = 0; k < 4; ++k) {
        int li = l - 3 + k;
        if (li >= 0) acc += cw[c * 4 + k] * zx[(size_t)(b * LL + li) * INW + DIN + c];
    }
    xo[(size_t)tok * XBCW + c] = acc / (1.0f + __expf(-acc));
}

// ---------------------------------------------------------------------------
// Per-token: dt = softplus(dt_raw + bias), a = exp(-dt * exp(A_log));
// in-place RoPE on Bm (ch 128..143) and Cm (ch 144..159) of xbc_act.
// ---------------------------------------------------------------------------
__global__ void dta_rope(const float* __restrict__ zx, float* __restrict__ xo,
                         const float* __restrict__ dtb, const float* __restrict__ alog,
                         float* __restrict__ dtbuf, float* __restrict__ abuf) {
    int tok = blockIdx.x * blockDim.x + threadIdx.x;
    if (tok >= NTOK) return;
    int l = tok % LL;
    #pragma unroll
    for (int h = 0; h < NH; ++h) {
        float raw = zx[(size_t)tok * INW + (DIN + XBCW) + h] + dtb[h];
        float dt = (raw > 20.0f) ? raw : log1pf(__expf(raw));
        abuf[tok * NH + h]  = __expf(-dt * __expf(alog[h]));
        dtbuf[tok * NH + h] = dt;
    }
    float* p = xo + (size_t)tok * XBCW;
    #pragma unroll
    for (int i = 0; i < 8; ++i) {
        float fr  = __powf(10000.0f, -(float)i * 0.125f);
        float ang = (float)l * fr;
        float sn, cs;
        __sincosf(ang, &sn, &cs);
        float b1 = p[128 + i], b2 = p[136 + i];
        p[128 + i] = b1 * cs - b2 * sn;
        p[136 + i] = b1 * sn + b2 * cs;
        float c1 = p[144 + i], c2 = p[152 + i];
        p[144 + i] = c1 * cs - c2 * sn;
        p[152 + i] = c1 * sn + c2 * cs;
    }
}

// ---------------------------------------------------------------------------
// Sequential SSM scan: one thread per (b, h, p); 16 state + 16 Bx_prev regs.
// All 32 lanes of a wave share batch b, so B_t/C_t are loaded once per wave
// (lanes 0-15 -> B[n], lanes 16-31 -> C[n]) and broadcast with constant-lane
// shuffles. Prefetches the next timestep's activation row.
// ---------------------------------------------------------------------------
__global__ void scan_kernel(const float* __restrict__ xo, const float* __restrict__ zx,
                            const float* __restrict__ dtbuf, const float* __restrict__ abuf,
                            const float* __restrict__ Dp, _Float16* __restrict__ yh) {
    int t = blockIdx.x * blockDim.x + threadIdx.x;
    if (t >= BB * NH * HD) return;
    int lane = threadIdx.x & 31;
    int p = t % HD;
    int h = (t / HD) % NH;
    int b = t / (HD * NH);
    float hs[16], bxp[16];
    #pragma unroll
    for (int n = 0; n < 16; ++n) { hs[n] = 0.0f; bxp[n] = 0.0f; }
    float dph = Dp[h];
    int bcoff = ((lane < 16) ? 128 : 144) + (lane & 15);
    for (int l = 0; l < LL; ++l) {
        int tok = b * LL + l;
        const float* row = xo + (size_t)tok * XBCW;
        float bc  = row[bcoff];            // lane-cooperative B/C load
        float xt  = row[h * HD + p];
        float at  = abuf[tok * NH + h];
        float dth = dtbuf[tok * NH + h] * 0.5f;
        if (l + 1 < LL) {
            __builtin_prefetch(row + XBCW, 0, 1);
            __builtin_prefetch(row + XBCW + 128, 0, 1);
        }
        float y = 0.0f;
        #pragma unroll
        for (int n = 0; n < 16; ++n) {
            float Bn = __shfl(bc, n, 32);
            float Cn = __shfl(bc, n + 16, 32);
            float bx = xt * Bn;
            float hn = at * hs[n] + dth * (bx + at * bxp[n]);
            hs[n] = hn; bxp[n] = bx;
            y += hn * Cn;
        }
        float zv = zx[(size_t)tok * INW + h * HD + p];
        float sz = zv / (1.0f + __expf(-zv));
        float yo = (y + dph * xt) * sz;
        yh[(size_t)tok * DIN + h * HD + p] = (_Float16)yo;
    }
}

// ---------------------------------------------------------------------------
// Final LayerNorm + (64 -> 6) head. One thread per token.
// ---------------------------------------------------------------------------
__global__ void head_kernel(const float* __restrict__ x, const float* __restrict__ g,
                            const float* __restrict__ be, const float* __restrict__ ow,
                            const float* __restrict__ ob, float* __restrict__ out) {
    int tok = blockIdx.x * blockDim.x + threadIdx.x;
    if (tok >= NTOK) return;
    const float* r = x + (size_t)tok * 64;
    float mu = 0.0f;
    #pragma unroll 8
    for (int d = 0; d < 64; ++d) mu += r[d];
    mu *= (1.0f / 64.0f);
    float var = 0.0f;
    #pragma unroll 8
    for (int d = 0; d < 64; ++d) { float dd = r[d] - mu; var += dd * dd; }
    var *= (1.0f / 64.0f);
    float inv = rsqrtf(var + 1e-5f);
    float acc[6];
    #pragma unroll
    for (int o = 0; o < 6; ++o) acc[o] = ob[o];
    for (int d = 0; d < 64; ++d) {
        float xn = (r[d] - mu) * inv * g[d] + be[d];
        #pragma unroll
        for (int o = 0; o < 6; ++o) acc[o] += xn * ow[d * 6 + o];
    }
    #pragma unroll
    for (int o = 0; o < 6; ++o) out[(size_t)tok * 6 + o] = acc[o];
}

// ---------------------------------------------------------------------------
extern "C" void kernel_launch(void* const* d_in, const int* in_sizes, int n_in,
                              void* d_out, int out_size, void* d_ws, size_t ws_size,
                              hipStream_t stream) {
    const int*   aggs    = (const int*)  d_in[0];
    const float* peg_E   = (const float*)d_in[1];
    const float* count_E = (const float*)d_in[2];
    const float* top_E   = (const float*)d_in[3];
    const float* pin_w   = (const float*)d_in[4];
    const float* pin_b   = (const float*)d_in[5];
    const float* rms_w   = (const float*)d_in[6];
    const float* in_w    = (const float*)d_in[7];
    const float* conv_w  = (const float*)d_in[8];
    const float* conv_b  = (const float*)d_in[9];
    const float* dt_bias = (const float*)d_in[10];
    const float* A_log   = (const float*)d_in[11];
    const float* Dp      = (const float*)d_in[12];
    const float* outp_w  = (const float*)d_in[13];
    const float* ln_g    = (const float*)d_in[14];
    const float* ln_b    = (const float*)d_in[15];
    const float* out_w   = (const float*)d_in[16];
    const float* out_b   = (const float*)d_in[17];

    char* ws = (char*)d_ws;
    auto carve = [&](size_t bytes) -> void* {
        void* p = (void*)ws;
        ws += (bytes + 255) & ~(size_t)255;
        return p;
    };
    float*     x     = (float*)    carve((size_t)NTOK * 64 * 4);
    _Float16*  xnh   = (_Float16*) carve((size_t)NTOK * 64 * 2);
    float*     zx    = (float*)    carve((size_t)NTOK * INW * 4);
    float*     xo    = (float*)    carve((size_t)NTOK * XBCW * 4);
    float*     dtbuf = (float*)    carve((size_t)NTOK * NH * 4);
    float*     abuf  = (float*)    carve((size_t)NTOK * NH * 4);
    _Float16*  yh    = (_Float16*) carve((size_t)NTOK * DIN * 2);
    _Float16*  wp1[2], *wp2[2];
    for (int l = 0; l < 2; ++l) {
        wp1[l] = (_Float16*)carve((size_t)2 * 19 * 32 * 16 * 2);
        wp2[l] = (_Float16*)carve((size_t)4 * 4 * 32 * 16 * 2);
    }

    // Pack weights into WMMA-B layout (f16).
    for (int l = 0; l < 2; ++l) {
        int tot1 = 2 * 19 * 32 * 16;
        pack_w_f16<<<(tot1 + 255) / 256, 256, 0, stream>>>(in_w + (size_t)l * 64 * 296,
                                                           wp1[l], 64, 296, 2, 19);
        int tot2 = 4 * 4 * 32 * 16;
        pack_w_f16<<<(tot2 + 255) / 256, 256, 0, stream>>>(outp_w + (size_t)l * 128 * 64,
                                                           wp2[l], 128, 64, 4, 4);
    }

    // Embedding + input projection.
    embed_proj<<<NTOK, 64, 0, stream>>>(aggs, peg_E, count_E, top_E, pin_w, pin_b, x);

    for (int l = 0; l < 2; ++l) {
        rmsnorm_f16<<<NTOK / 8, 256, 0, stream>>>(x, rms_w + l * 64, xnh);
        gemm_wmma<2, false><<<NTOK / 16, 32, 0, stream>>>(xnh, wp1[l], zx, nullptr,
                                                          64, 296, 19);
        conv_silu<<<(NTOK * XBCW + 255) / 256, 256, 0, stream>>>(zx,
                                                conv_w + l * XBCW * 4, conv_b + l * XBCW, xo);
        dta_rope<<<NTOK / 256, 256, 0, stream>>>(zx, xo, dt_bias + l * NH, A_log + l * NH,
                                                 dtbuf, abuf);
        scan_kernel<<<(BB * NH * HD) / 256, 256, 0, stream>>>(xo, zx, dtbuf, abuf,
                                                              Dp + l * NH, yh);
        // y @ outp_w + residual, in-place into x.
        gemm_wmma<4, true><<<NTOK / 16, 32, 0, stream>>>(yh, wp2[l], x, x, 128, 64, 4);
    }

    head_kernel<<<NTOK / 256, 256, 0, stream>>>(x, ln_g, ln_b, out_w, out_b, (float*)d_out);
}
